// KBGraphAttentionNetwork_71459665871398
// MI455X (gfx1250) — matrix operand order repaired and checked by reading, not compiled, not run on hardware
//
#include <hip/hip_runtime.h>

// ---------------------------------------------------------------------------
// Types for CDNA5 WMMA (wave32, 16x16x32 bf16 -> f32)
// ---------------------------------------------------------------------------
typedef __bf16 v16bf __attribute__((ext_vector_type(16)));
typedef float  v8f   __attribute__((ext_vector_type(8)));

union Frag {
    v16bf v;
    uint4 q[2];
};

// ---------------------------------------------------------------------------
// Helpers
// ---------------------------------------------------------------------------
__device__ __forceinline__ unsigned short f2bf(float f) {
    unsigned u = __float_as_uint(f);
    u += 0x7FFFu + ((u >> 16) & 1u);   // round-to-nearest-even
    return (unsigned short)(u >> 16);
}
__device__ __forceinline__ float bf2f(unsigned short s) {
    return __uint_as_float(((unsigned)s) << 16);
}
// order-preserving float -> uint encoding (for atomicMax-based segment max)
__device__ __forceinline__ unsigned enc_ord(float x) {
    unsigned u = __float_as_uint(x);
    return (u & 0x80000000u) ? ~u : (u | 0x80000000u);
}
__device__ __forceinline__ float dec_ord(unsigned u) {
    unsigned b = (u & 0x80000000u) ? (u ^ 0x80000000u) : ~u;
    return __uint_as_float(b);
}
__device__ __forceinline__ float mish_f(float x) {
    float sp = (x > 20.f) ? x : log1pf(expf(x));
    return x * tanhf(sp);
}

// ---------------------------------------------------------------------------
// Convert f32 -> bf16
// ---------------------------------------------------------------------------
__global__ void cvt_bf16_kernel(const float* __restrict__ src,
                                unsigned short* __restrict__ dst, int n) {
    int t = blockIdx.x * 256 + threadIdx.x;
    if (t < n) dst[t] = f2bf(src[t]);
}

__global__ void fill_u32_kernel(unsigned* __restrict__ p, unsigned v, int n) {
    int t = blockIdx.x * 256 + threadIdx.x;
    if (t < n) p[t] = v;
}

// ---------------------------------------------------------------------------
// Gather-GEMM: h[e, o] = sum_f [ent[head]||ent[tail]||rel[r]][f] * W[o, f]
//   M-tile: 32 edges per workgroup (two 16-row WMMA subtiles -> each B
//   fragment feeds two WMMAs, halving L2 B-traffic).
//   A: 32 edges x 768 (bf16, staged in LDS, stride 776 to avoid bank conflicts)
//   B: W row-major (OUT x 768) bf16
//   OUT = NT * 8 waves * 16 = NT*128  (NT=2 -> 256, NT=8 -> 1024)
//   Output h stored bf16.
// ---------------------------------------------------------------------------
template <int NT>
__global__ __launch_bounds__(256)
void gat_gemm_kernel(const unsigned short* __restrict__ entb,
                     const unsigned short* __restrict__ relb,
                     const int* __restrict__ triples,
                     const unsigned short* __restrict__ Wb,
                     unsigned short* __restrict__ hout, int E_) {
    constexpr int OUT = NT * 128;
    __shared__ __align__(16) unsigned short smem[32 * 776];
    const int e0 = blockIdx.x * 32;
    const int tid = threadIdx.x;

    // Stage the 32x768 concat tile (chunks of 8 bf16 = 16B)
    for (int c = tid; c < 32 * 96; c += 256) {
        const int row = c / 96;
        const int col = (c % 96) * 8;
        int ee = e0 + row;
        if (ee >= E_) ee = E_ - 1;
        const int* tr = triples + (size_t)ee * 3;
        const unsigned short* src;
        if (col < 256)      src = entb + (size_t)tr[0] * 256 + col;
        else if (col < 512) src = entb + (size_t)tr[1] * 256 + (col - 256);
        else                src = relb + (size_t)tr[2] * 256 + (col - 512);
        *(uint4*)&smem[row * 776 + col] = *(const uint4*)src;
    }
    __syncthreads();

    const int wave = tid >> 5;
    const int lane = tid & 31;
    const int half = lane >> 4;    // lanes 0-15 vs 16-31
    const int mrow = lane & 15;    // M row for A, N col for B/C

    v8f acc[2][NT];
#pragma unroll
    for (int s = 0; s < 2; ++s)
#pragma unroll
        for (int t = 0; t < NT; ++t)
#pragma unroll
            for (int j = 0; j < 8; ++j) acc[s][t][j] = 0.f;

    for (int ks = 0; ks < 24; ++ks) {
        const int k0 = ks * 32;
        // A fragments (ISA 16-bit A 16x32 layout): lane(m,half),
        // elems 0-7 -> K = k0+half*8 .. +7 ; elems 8-15 -> K = k0+16+half*8 .. +7
        Frag A0, A1;
        A0.q[0] = *(const uint4*)&smem[mrow * 776 + k0 + half * 8];
        A0.q[1] = *(const uint4*)&smem[mrow * 776 + k0 + 16 + half * 8];
        A1.q[0] = *(const uint4*)&smem[(mrow + 16) * 776 + k0 + half * 8];
        A1.q[1] = *(const uint4*)&smem[(mrow + 16) * 776 + k0 + 16 + half * 8];
#pragma unroll
        for (int t = 0; t < NT; ++t) {
            const int n = (wave * NT + t) * 16 + mrow;
            // B fragment: lane holds column n, elems 0-7 -> K = k0+half*8.., elems 8-15 -> K = k0+16+half*8..
            const unsigned short* bp = Wb + (size_t)n * 768 + k0 + half * 8;
            Frag B;
            B.q[0] = ((const uint4*)bp)[0];
            B.q[1] = ((const uint4*)bp)[2];
            acc[0][t] = __builtin_amdgcn_wmma_f32_16x16x32_bf16(
                false, A0.v, false, B.v, (short)0, acc[0][t], false, false);
            acc[1][t] = __builtin_amdgcn_wmma_f32_16x16x32_bf16(
                false, A1.v, false, B.v, (short)0, acc[1][t], false, false);
        }
    }

    // C layout: elem j -> (M = half*8 + j, N = lane%16)
#pragma unroll
    for (int s = 0; s < 2; ++s)
#pragma unroll
        for (int t = 0; t < NT; ++t) {
            const int n = (wave * NT + t) * 16 + mrow;
#pragma unroll
            for (int j = 0; j < 8; ++j) {
                const int m = s * 16 + half * 8 + j;
                if (e0 + m < E_)
                    hout[(size_t)(e0 + m) * OUT + n] = f2bf(acc[s][t][j]);
            }
        }
}

// ---------------------------------------------------------------------------
// Dense GEMM + bias: out[m, n] = sum_k A[m,k]*W[n,k] + bias[n]; K = 256.
// M-tile 32 (two 16-row subtiles sharing each B fragment).
// ---------------------------------------------------------------------------
__global__ __launch_bounds__(256)
void dense_gemm_bias_kernel(const unsigned short* __restrict__ Ab,
                            const unsigned short* __restrict__ Wb,
                            const float* __restrict__ bias,
                            float* __restrict__ out, int rows) {
    __shared__ __align__(16) unsigned short smem[32 * 264];
    const int e0 = blockIdx.x * 32;
    const int tid = threadIdx.x;
    for (int c = tid; c < 32 * 32; c += 256) {
        const int row = c / 32;
        const int col = (c % 32) * 8;
        int rr = e0 + row;
        if (rr >= rows) rr = rows - 1;
        *(uint4*)&smem[row * 264 + col] = *(const uint4*)(Ab + (size_t)rr * 256 + col);
    }
    __syncthreads();

    const int wave = tid >> 5, lane = tid & 31;
    const int half = lane >> 4, mrow = lane & 15;

    v8f acc[2][2];
#pragma unroll
    for (int s = 0; s < 2; ++s)
#pragma unroll
        for (int t = 0; t < 2; ++t)
#pragma unroll
            for (int j = 0; j < 8; ++j) acc[s][t][j] = 0.f;

    for (int ks = 0; ks < 8; ++ks) {
        const int k0 = ks * 32;
        Frag A0, A1;
        A0.q[0] = *(const uint4*)&smem[mrow * 264 + k0 + half * 8];
        A0.q[1] = *(const uint4*)&smem[mrow * 264 + k0 + 16 + half * 8];
        A1.q[0] = *(const uint4*)&smem[(mrow + 16) * 264 + k0 + half * 8];
        A1.q[1] = *(const uint4*)&smem[(mrow + 16) * 264 + k0 + 16 + half * 8];
#pragma unroll
        for (int t = 0; t < 2; ++t) {
            const int n = (wave * 2 + t) * 16 + mrow;
            const unsigned short* bp = Wb + (size_t)n * 256 + k0 + half * 8;
            Frag B;
            B.q[0] = ((const uint4*)bp)[0];
            B.q[1] = ((const uint4*)bp)[2];
            acc[0][t] = __builtin_amdgcn_wmma_f32_16x16x32_bf16(
                false, A0.v, false, B.v, (short)0, acc[0][t], false, false);
            acc[1][t] = __builtin_amdgcn_wmma_f32_16x16x32_bf16(
                false, A1.v, false, B.v, (short)0, acc[1][t], false, false);
        }
    }
#pragma unroll
    for (int s = 0; s < 2; ++s)
#pragma unroll
        for (int t = 0; t < 2; ++t) {
            const int n = (wave * 2 + t) * 16 + mrow;
            const float bv = bias[n];
#pragma unroll
            for (int j = 0; j < 8; ++j) {
                const int m = s * 16 + half * 8 + j;
                if (e0 + m < rows)
                    out[(size_t)(e0 + m) * 256 + n] = acc[s][t][j] + bv;
            }
        }
}

// ---------------------------------------------------------------------------
// Attention score: a[e,h] = mish( h[e,h,:] . W_a[h,:] ), + segment max
// ---------------------------------------------------------------------------
__global__ void attn_a_kernel(const unsigned short* __restrict__ hb,
                              const float* __restrict__ Wa,
                              const int* __restrict__ triples,
                              float* __restrict__ abuf,
                              unsigned* __restrict__ mseg,
                              int OUT, int DH, int E_) {
    int t = blockIdx.x * 256 + threadIdx.x;
    if (t >= E_ * 4) return;
    const int e = t >> 2, hh = t & 3;
    const unsigned short* hp = hb + (size_t)e * OUT + hh * DH;
    const float* wp = Wa + hh * DH;
    float s = 0.f;
    for (int d = 0; d < DH; ++d) s += bf2f(hp[d]) * wp[d];
    const float a = mish_f(s);
    abuf[t] = a;
    const int seg = triples[(size_t)e * 3];
    atomicMax(&mseg[seg * 4 + hh], enc_ord(a));
}

// ex = exp(a - m[seg]); den[seg] += ex  (a overwritten by ex)
__global__ void attn_ex_kernel(float* __restrict__ abuf,
                               const unsigned* __restrict__ mseg,
                               float* __restrict__ dseg,
                               const int* __restrict__ triples, int E_) {
    int t = blockIdx.x * 256 + threadIdx.x;
    if (t >= E_ * 4) return;
    const int e = t >> 2, hh = t & 3;
    const int seg = triples[(size_t)e * 3];
    const float m = dec_ord(mseg[seg * 4 + hh]);
    const float ex = expf(abuf[t] - m);
    abuf[t] = ex;
    atomicAdd(&dseg[seg * 4 + hh], ex);
}

// Weighted scatter. MODE 0: concat heads (OUT=256, col c -> head c/64).
// MODE 1: mean over heads (OUT=1024, col d summed over 4 heads, *0.25).
template <int MODE>
__global__ void scatter_kernel(const unsigned short* __restrict__ hb,
                               const float* __restrict__ exbuf,
                               const float* __restrict__ dseg,
                               const int* __restrict__ triples,
                               float* __restrict__ out, int E_) {
    int e = blockIdx.x;
    if (e >= E_) return;
    const int c = threadIdx.x;      // 0..255
    const int seg = triples[(size_t)e * 3];
    float val;
    if (MODE == 0) {
        const int hh = c >> 6;
        const float alpha = exbuf[e * 4 + hh] / dseg[seg * 4 + hh];
        val = alpha * bf2f(hb[(size_t)e * 256 + c]);
    } else {
        float s = 0.f;
#pragma unroll
        for (int hh = 0; hh < 4; ++hh) {
            const float alpha = exbuf[e * 4 + hh] / dseg[seg * 4 + hh];
            s += alpha * bf2f(hb[(size_t)e * 1024 + hh * 256 + c]);
        }
        val = s * 0.25f;
    }
    atomicAdd(&out[(size_t)seg * 256 + c], val);
}

// ---------------------------------------------------------------------------
// Small f32 linear: out[i,j] = sum_k A[i,k]*B[j,k] (+ bias[j])   (K = 256)
// ---------------------------------------------------------------------------
__global__ void lin_kernel(const float* __restrict__ A, const float* __restrict__ B,
                           const float* __restrict__ bias, float* __restrict__ out,
                           int rows) {
    int t = blockIdx.x * 256 + threadIdx.x;
    if (t >= rows * 256) return;
    const int i = t >> 8, j = t & 255;
    const float* ap = A + (size_t)i * 256;
    const float* bp = B + (size_t)j * 256;
    float s = bias ? bias[j] : 0.f;
    for (int k = 0; k < 256; ++k) s += ap[k] * bp[k];
    out[t] = s;
}

// ---------------------------------------------------------------------------
// Residual + LayerNorm (D=256). One wave32 per row; writes f32 + bf16 copies.
// ---------------------------------------------------------------------------
__global__ __launch_bounds__(256)
void resid_ln_kernel(const float* __restrict__ base, const float* __restrict__ delta,
                     const float* __restrict__ g, const float* __restrict__ b,
                     float* __restrict__ y32, unsigned short* __restrict__ ybf,
                     int rows) {
    const int wave = threadIdx.x >> 5, lane = threadIdx.x & 31;
    const int n = blockIdx.x * 8 + wave;
    if (n >= rows) return;
    const float* bp = base + (size_t)n * 256;
    const float* dp = delta + (size_t)n * 256;
    float x[8], s = 0.f, s2 = 0.f;
#pragma unroll
    for (int i = 0; i < 8; ++i) {
        const int c = lane + i * 32;
        x[i] = bp[c] + dp[c];
        s += x[i];
        s2 += x[i] * x[i];
    }
    for (int off = 16; off > 0; off >>= 1) {
        s  += __shfl_xor(s, off, 32);
        s2 += __shfl_xor(s2, off, 32);
    }
    const float mean = s * (1.f / 256.f);
    const float var = s2 * (1.f / 256.f) - mean * mean;
    const float inv = rsqrtf(var + 1e-5f);
#pragma unroll
    for (int i = 0; i < 8; ++i) {
        const int c = lane + i * 32;
        const float y = (x[i] - mean) * inv * g[c] + b[c];
        y32[(size_t)n * 256 + c] = y;
        ybf[(size_t)n * 256 + c] = f2bf(y);
    }
}

// ---------------------------------------------------------------------------
// Host orchestration
// ---------------------------------------------------------------------------
extern "C" void kernel_launch(void* const* d_in, const int* in_sizes, int n_in,
                              void* d_out, int out_size, void* d_ws, size_t ws_size,
                              hipStream_t stream) {
    const float* ent   = (const float*)d_in[0];
    const float* rel   = (const float*)d_in[1];
    const int*   trip  = (const int*)d_in[2];
    const float* Wt1   = (const float*)d_in[3];
    const float* Wa1   = (const float*)d_in[4];
    const float* Wrel1 = (const float*)d_in[5];
    const float* ln1eg = (const float*)d_in[6];
    const float* ln1eb = (const float*)d_in[7];
    const float* ln1rg = (const float*)d_in[8];
    const float* ln1rb = (const float*)d_in[9];
    const float* Wt2   = (const float*)d_in[10];
    const float* Wa2   = (const float*)d_in[11];
    const float* Wrel2 = (const float*)d_in[12];
    const float* ln2eg = (const float*)d_in[13];
    const float* ln2eb = (const float*)d_in[14];
    const float* ln2rg = (const float*)d_in[15];
    const float* ln2rb = (const float*)d_in[16];
    const float* Went  = (const float*)d_in[17];
    const float* bent  = (const float*)d_in[18];
    const float* Wrelf = (const float*)d_in[19];
    const float* brelf = (const float*)d_in[20];

    const int NE = in_sizes[0] / 256;   // 20000
    const int NR = in_sizes[1] / 256;   // 200
    const int E  = in_sizes[2] / 3;     // 100000

    // ---- workspace layout (256B aligned bump allocator) ----
    char* ws = (char*)d_ws;
    size_t off = 0;
    auto alloc = [&](size_t bytes) -> char* {
        char* p = ws + off;
        off = (off + bytes + 255) & ~(size_t)255;
        return p;
    };
    unsigned short* entbf  = (unsigned short*)alloc((size_t)NE * 256 * 2);
    unsigned short* relbf  = (unsigned short*)alloc((size_t)NR * 256 * 2);
    unsigned short* wt1bf  = (unsigned short*)alloc((size_t)256 * 768 * 2);
    unsigned short* wt2bf  = (unsigned short*)alloc((size_t)1024 * 768 * 2);
    unsigned short* wentbf = (unsigned short*)alloc((size_t)256 * 256 * 2);
    unsigned short* hbuf   = (unsigned short*)alloc((size_t)E * 1024 * 2);  // shared by both layers
    float*    abuf  = (float*)alloc((size_t)E * 4 * 4);
    unsigned* mseg  = (unsigned*)alloc((size_t)NE * 4 * 4);
    float*    dseg  = (float*)alloc((size_t)NE * 4 * 4);
    float*    outsc = (float*)alloc((size_t)NE * 256 * 4);  // shared by both layers
    float*    e1    = (float*)alloc((size_t)NE * 256 * 4);
    unsigned short* e1bf = (unsigned short*)alloc((size_t)NE * 256 * 2);
    float*    rlin  = (float*)alloc((size_t)NR * 256 * 4);  // shared by both layers
    float*    r1    = (float*)alloc((size_t)NR * 256 * 4);
    unsigned short* r1bf = (unsigned short*)alloc((size_t)NR * 256 * 2);
    float*    e2    = (float*)alloc((size_t)NE * 256 * 4);
    unsigned short* e2bf = (unsigned short*)alloc((size_t)NE * 256 * 2);
    float*    r2    = (float*)alloc((size_t)NR * 256 * 4);
    unsigned short* r2bf = (unsigned short*)alloc((size_t)NR * 256 * 2);
    (void)ws_size; (void)n_in;

    auto cg = [](long long n) { return dim3((unsigned)((n + 255) / 256)); };

    // ---- precision conversion (f32 -> bf16) ----
    cvt_bf16_kernel<<<cg((long long)NE * 256), 256, 0, stream>>>(ent, entbf, NE * 256);
    cvt_bf16_kernel<<<cg((long long)NR * 256), 256, 0, stream>>>(rel, relbf, NR * 256);
    cvt_bf16_kernel<<<cg(256 * 768), 256, 0, stream>>>(Wt1, wt1bf, 256 * 768);
    cvt_bf16_kernel<<<cg(1024 * 768), 256, 0, stream>>>(Wt2, wt2bf, 1024 * 768);
    cvt_bf16_kernel<<<cg(256 * 256), 256, 0, stream>>>(Went, wentbf, 256 * 256);

    const dim3 gE32((unsigned)((E + 31) / 32));

    // ================= Layer 1 =================
    fill_u32_kernel<<<cg(NE * 4), 256, 0, stream>>>(mseg, 0u, NE * 4);
    fill_u32_kernel<<<cg(NE * 4), 256, 0, stream>>>((unsigned*)dseg, 0u, NE * 4);
    fill_u32_kernel<<<cg((long long)NE * 256), 256, 0, stream>>>((unsigned*)outsc, 0u, NE * 256);

    gat_gemm_kernel<2><<<gE32, 256, 0, stream>>>(entbf, relbf, trip, wt1bf, hbuf, E);
    attn_a_kernel<<<cg((long long)E * 4), 256, 0, stream>>>(hbuf, Wa1, trip, abuf, mseg, 256, 64, E);
    attn_ex_kernel<<<cg((long long)E * 4), 256, 0, stream>>>(abuf, mseg, dseg, trip, E);
    scatter_kernel<0><<<dim3(E), 256, 0, stream>>>(hbuf, abuf, dseg, trip, outsc, E);

    lin_kernel<<<cg((long long)NR * 256), 256, 0, stream>>>(rel, Wrel1, nullptr, rlin, NR);
    resid_ln_kernel<<<dim3((NE + 7) / 8), 256, 0, stream>>>(ent, outsc, ln1eg, ln1eb, e1, e1bf, NE);
    resid_ln_kernel<<<dim3((NR + 7) / 8), 256, 0, stream>>>(rel, rlin, ln1rg, ln1rb, r1, r1bf, NR);

    // ================= Layer 2 =================
    fill_u32_kernel<<<cg(NE * 4), 256, 0, stream>>>(mseg, 0u, NE * 4);
    fill_u32_kernel<<<cg(NE * 4), 256, 0, stream>>>((unsigned*)dseg, 0u, NE * 4);
    fill_u32_kernel<<<cg((long long)NE * 256), 256, 0, stream>>>((unsigned*)outsc, 0u, NE * 256);

    gat_gemm_kernel<8><<<gE32, 256, 0, stream>>>(e1bf, r1bf, trip, wt2bf, hbuf, E);
    attn_a_kernel<<<cg((long long)E * 4), 256, 0, stream>>>(hbuf, Wa2, trip, abuf, mseg, 1024, 256, E);
    attn_ex_kernel<<<cg((long long)E * 4), 256, 0, stream>>>(abuf, mseg, dseg, trip, E);
    scatter_kernel<1><<<dim3(E), 256, 0, stream>>>(hbuf, abuf, dseg, trip, outsc, E);

    lin_kernel<<<cg((long long)NR * 256), 256, 0, stream>>>(r1, Wrel2, nullptr, rlin, NR);
    resid_ln_kernel<<<dim3((NE + 7) / 8), 256, 0, stream>>>(e1, outsc, ln2eg, ln2eb, e2, e2bf, NE);
    resid_ln_kernel<<<dim3((NR + 7) / 8), 256, 0, stream>>>(r1, rlin, ln2rg, ln2rb, r2, r2bf, NR);

    // ================= Final projections =================
    float* eout = (float*)d_out;                       // (NE, 256)
    float* rout = (float*)d_out + (size_t)NE * 256;    // (NR, 256)
    dense_gemm_bias_kernel<<<dim3((NE + 31) / 32), 256, 0, stream>>>(e2bf, wentbf, bent, eout, NE);
    lin_kernel<<<cg((long long)NR * 256), 256, 0, stream>>>(r2, Wrelf, brelf, rout, NR);
}
